// GNNEncoder_41927470744186
// MI455X (gfx1250) — matrix operand order repaired and checked by reading, not compiled, not run on hardware
//
#include <hip/hip_runtime.h>
#include <math.h>

// ---------------------------------------------------------------------------
// Problem constants (from the reference)
// ---------------------------------------------------------------------------
constexpr int N_DC = 100, N_HOST = 100000, N_VM = 500000,
              N_TASK = 1000000, N_INST = 500000;
constexpr int E_DSHO = 200000, E_DSVM = 500000, E_HODS = 200000,
              E_VMDS = 500000, E_DEP = 4000000, E_PART = 2000000,
              E_RUNIN = 1000000, E_RUNBY = 1000000;
constexpr int IN_DIM = 16, HID = 8, EMB = 5;
constexpr float NEG_SLOPE = 0.2f;
constexpr int TILES = 4;   // 16-row WMMA tiles per wave
constexpr int WPB   = 4;   // waves per block (128 threads)

typedef __attribute__((ext_vector_type(16))) _Float16 v16h;
typedef __attribute__((ext_vector_type(8)))  float    v8f;

// ---------------------------------------------------------------------------
// WMMA GEMM:  out[N x DOUT] = X[N x DIN] @ W[DIN x DOUT] (+bias1 +bias2)
// One wave computes TILES 16x16 tiles via v_wmma_f32_16x16x32_f16 (K padded
// to 32). DIN in {16,8}, DOUT in {8,5}. Branchless vector loads (row index
// clamped; stores guarded). Full tiles are transposed through per-wave LDS
// so all 32 lanes issue one coalesced global_store_b128 per tile.
// ---------------------------------------------------------------------------
template <int DIN, int DOUT, bool ACC>
__global__ void gnn_gemm_wmma(const float* __restrict__ X,
                              const float* __restrict__ Wm,
                              const float* __restrict__ b1,
                              const float* __restrict__ b2,
                              float* __restrict__ out, int N)
{
  __shared__ __align__(16) float stage[WPB][16 * DOUT];

  const int lane = threadIdx.x & 31;
  const int wid  = threadIdx.x >> 5;
  const int wave = blockIdx.x * WPB + wid;
  const int row0 = wave * 16 * TILES;
  if (row0 >= N) return;              // wave-uniform: EXEC all-ones below

  const int half = lane >> 4;         // 0: lanes 0-15, 1: lanes 16-31
  const int mr   = lane & 15;

  // ---- B fragment (32x16 f16): lane holds column N=mr; lanes 0-15 cover
  // K=0..7 then 16..23, lanes 16-31 K=8..15 then 24..31. Clamped indices + select: no branches. ----
  v16h b;
  #pragma unroll
  for (int i = 0; i < 16; ++i) {
    const int k  = (i < 8) ? half * 8 + i : 16 + half * 8 + (i - 8);
    const int kc = (k < DIN) ? k : 0;
    const int cc = (mr < DOUT) ? mr : 0;
    const float v = Wm[kc * DOUT + cc];
    b[i] = (_Float16)(((k < DIN) && (mr < DOUT)) ? v : 0.0f);
  }

  float bias = 0.0f;
  if (mr < DOUT) {
    if (b1) bias += b1[mr];
    if (b2) bias += b2[mr];
  }

  #pragma unroll
  for (int t = 0; t < TILES; ++t) {
    const int r0 = row0 + t * 16;
    if (r0 >= N) break;               // wave-uniform

    // ---- A fragment (16x32 f16): lane holds row M=mr.
    // lanes 0-15: K=0..7 then 16..23 ; lanes 16-31: K=8..15 then 24..31 ----
    const int arow = r0 + mr;
    const int arc  = (arow < N) ? arow : (N - 1);   // clamp; dup rows unused
    v16h a;
    #pragma unroll
    for (int i = 8; i < 16; ++i) a[i] = (_Float16)0.0f;  // K >= 16 always 0
    if constexpr (DIN == 16) {
      const float4* px = (const float4*)(X + (size_t)arc * 16 + half * 8);
      const float4 q0 = px[0], q1 = px[1];
      a[0] = (_Float16)q0.x; a[1] = (_Float16)q0.y;
      a[2] = (_Float16)q0.z; a[3] = (_Float16)q0.w;
      a[4] = (_Float16)q1.x; a[5] = (_Float16)q1.y;
      a[6] = (_Float16)q1.z; a[7] = (_Float16)q1.w;
    } else {                          // DIN == 8: lanes 16-31 hold K>=8 -> 0
      const float4* px = (const float4*)(X + (size_t)arc * 8);
      const float4 q0 = px[0], q1 = px[1];
      const float m = (half == 0) ? 1.0f : 0.0f;
      a[0] = (_Float16)(q0.x * m); a[1] = (_Float16)(q0.y * m);
      a[2] = (_Float16)(q0.z * m); a[3] = (_Float16)(q0.w * m);
      a[4] = (_Float16)(q1.x * m); a[5] = (_Float16)(q1.y * m);
      a[6] = (_Float16)(q1.z * m); a[7] = (_Float16)(q1.w * m);
    }

    v8f c = {};
    c = __builtin_amdgcn_wmma_f32_16x16x32_f16(
        false, a, false, b, (short)0, c, false, false);

    // ---- D store: lanes 0-15 hold rows r0..r0+7 (VGPR r -> M=r), lanes
    // 16-31 rows r0+8..r0+15; col = mr. Full tiles go through LDS so every
    // lane stores one contiguous float4 (coalesced b128). ----
    if (r0 + 16 <= N) {               // full tile (r0 multiple of 16)
      float* st = stage[wid];
      if (mr < DOUT) {
        #pragma unroll
        for (int r = 0; r < 8; ++r)
          st[(half * 8 + r) * DOUT + mr] = c[r] + bias;
      }
      // intra-wave LDS produce->consume: DS ops are in-order per wave;
      // compiler inserts the s_wait_dscnt.
      constexpr int TOT = 16 * DOUT;  // floats in tile (128 or 80)
      const int l4 = lane * 4;
      if (l4 < TOT) {
        const float4 v = *(const float4*)(st + l4);
        float4* gp = (float4*)(out + (size_t)r0 * DOUT + l4);
        if (ACC) {
          float4 o = *gp;
          o.x += v.x; o.y += v.y; o.z += v.z; o.w += v.w;
          *gp = o;
        } else {
          *gp = v;
        }
      }
    } else {                          // tail tile: guarded scalar stores
      if (mr < DOUT) {
        float* op = out + (size_t)(r0 + half * 8) * DOUT + mr;
        #pragma unroll
        for (int r = 0; r < 8; ++r) {
          if (r0 + half * 8 + r < N) {
            const float v = c[r] + bias;
            if (ACC) op[(size_t)r * DOUT] += v; else op[(size_t)r * DOUT] = v;
          }
        }
      }
    }
  }
}

// ---------------------------------------------------------------------------
// Scalar edge / utility kernels (L2-resident atomic scatters, b128 row loads)
// ---------------------------------------------------------------------------
__global__ void gnn_fill(float* p, float v, int n) {
  int i = blockIdx.x * blockDim.x + threadIdx.x;
  if (i < n) p[i] = v;
}

__global__ void gnn_fill_bias8(float* p, const float* __restrict__ b, int n_nodes) {
  int i = blockIdx.x * blockDim.x + threadIdx.x;
  if (i < n_nodes * HID) p[i] = b[i & 7];
}

__global__ void gnn_relu(float* p, int n) {
  int i = blockIdx.x * blockDim.x + threadIdx.x;
  if (i < n) p[i] = fmaxf(p[i], 0.0f);
}

__global__ void gnn_scatter_add(const float* __restrict__ h,
                                const int* __restrict__ es,
                                const int* __restrict__ ed,
                                float* acc, int E) {
  int e = blockIdx.x * blockDim.x + threadIdx.x;
  if (e >= E) return;
  const float4* hs = (const float4*)(h + (size_t)es[e] * HID);
  const float4 lo = hs[0], hi = hs[1];
  float* a = acc + (size_t)ed[e] * HID;
  atomicAdd(&a[0], lo.x); atomicAdd(&a[1], lo.y);
  atomicAdd(&a[2], lo.z); atomicAdd(&a[3], lo.w);
  atomicAdd(&a[4], hi.x); atomicAdd(&a[5], hi.y);
  atomicAdd(&a[6], hi.z); atomicAdd(&a[7], hi.w);
}

__global__ void gnn_deg_count(const int* __restrict__ ed, float* deg, int E) {
  int e = blockIdx.x * blockDim.x + threadIdx.x;
  if (e < E) atomicAdd(&deg[ed[e]], 1.0f);
}

__global__ void gnn_deg_inv(float* deg, int n) {
  int i = blockIdx.x * blockDim.x + threadIdx.x;
  if (i >= n) return;
  float d = deg[i];
  deg[i] = (d > 0.0f) ? rsqrtf(fmaxf(d, 1.0f)) : 0.0f;
}

__global__ void gnn_gcn_scatter(const float* __restrict__ h,
                                const float* __restrict__ dinv,
                                const int* __restrict__ es,
                                const int* __restrict__ ed,
                                float* acc, int E) {
  int e = blockIdx.x * blockDim.x + threadIdx.x;
  if (e >= E) return;
  const int s = es[e], d0 = ed[e];
  const float norm = dinv[s] * dinv[d0];
  const float4* hs = (const float4*)(h + (size_t)s * HID);
  const float4 lo = hs[0], hi = hs[1];
  float* a = acc + (size_t)d0 * HID;
  atomicAdd(&a[0], norm * lo.x); atomicAdd(&a[1], norm * lo.y);
  atomicAdd(&a[2], norm * lo.z); atomicAdd(&a[3], norm * lo.w);
  atomicAdd(&a[4], norm * hi.x); atomicAdd(&a[5], norm * hi.y);
  atomicAdd(&a[6], norm * hi.z); atomicAdd(&a[7], norm * hi.w);
}

__device__ __forceinline__ void atomicMaxF(float* addr, float val) {
  // Monotone int-bit trick; buffer is initialized to -inf.
  if (val >= 0.0f) atomicMax((int*)addr, __float_as_int(val));
  else             atomicMin((unsigned int*)addr, (unsigned int)__float_as_int(val));
}

__global__ void gnn_gat_logit(const float* __restrict__ hl,
                              const float* __restrict__ hr,
                              const float* __restrict__ att,
                              const int* __restrict__ es,
                              const int* __restrict__ ed,
                              float* evals, float* mmax, int E) {
  int e = blockIdx.x * blockDim.x + threadIdx.x;
  if (e >= E) return;
  const int s = es[e], d0 = ed[e];
  const float4* pa = (const float4*)(hl + (size_t)s * HID);
  const float4* pb = (const float4*)(hr + (size_t)d0 * HID);
  const float4 a0 = pa[0], a1 = pa[1], b0 = pb[0], b1 = pb[1];
  float sv[HID] = {a0.x + b0.x, a0.y + b0.y, a0.z + b0.z, a0.w + b0.w,
                   a1.x + b1.x, a1.y + b1.y, a1.z + b1.z, a1.w + b1.w};
  float acc = 0.0f;
  #pragma unroll
  for (int c = 0; c < HID; ++c) {
    const float v = (sv[c] > 0.0f) ? sv[c] : NEG_SLOPE * sv[c];
    acc += v * att[c];
  }
  evals[e] = acc;
  atomicMaxF(&mmax[d0], acc);
}

__global__ void gnn_gat_exp(float* evals, const float* __restrict__ mmax,
                            float* den, const int* __restrict__ ed, int E) {
  int e = blockIdx.x * blockDim.x + threadIdx.x;
  if (e >= E) return;
  const float w = expf(evals[e] - mmax[ed[e]]);
  evals[e] = w;
  atomicAdd(&den[ed[e]], w);
}

__global__ void gnn_gat_scatter(const float* __restrict__ hl,
                                const float* __restrict__ evals,
                                const float* __restrict__ den,
                                const int* __restrict__ es,
                                const int* __restrict__ ed,
                                float* acc, int E) {
  int e = blockIdx.x * blockDim.x + threadIdx.x;
  if (e >= E) return;
  const int s = es[e], d0 = ed[e];
  const float alpha = evals[e] / (den[d0] + 1e-16f);
  const float4* hs = (const float4*)(hl + (size_t)s * HID);
  const float4 lo = hs[0], hi = hs[1];
  float* a = acc + (size_t)d0 * HID;
  atomicAdd(&a[0], alpha * lo.x); atomicAdd(&a[1], alpha * lo.y);
  atomicAdd(&a[2], alpha * lo.z); atomicAdd(&a[3], alpha * lo.w);
  atomicAdd(&a[4], alpha * hi.x); atomicAdd(&a[5], alpha * hi.y);
  atomicAdd(&a[6], alpha * hi.z); atomicAdd(&a[7], alpha * hi.w);
}

// ---------------------------------------------------------------------------
// Host-side parameter plumbing (JAX pytree order: dict keys sorted)
// ---------------------------------------------------------------------------
struct GcpP { const float *b_rel, *w_rel, *w_root; };
struct GatP { const float *att, *b, *bl, *br, *wl, *wr; };
struct GcnP { const float *b, *w; };
struct LayerP { GcnP depend; GcpP dsho, dsvm, hods; GatP part_of, run_by; GcpP run_in, vmds; };

struct ParamCursor {
  void* const* d_in; int idx; const float* blob; size_t off; bool per_leaf;
  const float* next(size_t n) {
    if (per_leaf) return (const float*)d_in[idx++];
    const float* p = blob + off; off += n; return p;
  }
};

static void read_layer(ParamCursor& c, LayerP& L, int d) {
  // sorted keys: depend, dsho, dsvm, hods, part_of, run_by, run_in, vmds
  L.depend.b = c.next(HID); L.depend.w = c.next((size_t)d * HID);
  auto gcp = [&](GcpP& g) {  // sorted: b_rel, w_rel, w_root
    g.b_rel = c.next(HID); g.w_rel = c.next((size_t)d * HID); g.w_root = c.next((size_t)d * HID);
  };
  auto gat = [&](GatP& g) {  // sorted: att, b, bl, br, wl, wr
    g.att = c.next(HID); g.b = c.next(HID); g.bl = c.next(HID); g.br = c.next(HID);
    g.wl = c.next((size_t)d * HID); g.wr = c.next((size_t)d * HID);
  };
  gcp(L.dsho); gcp(L.dsvm); gcp(L.hods);
  gat(L.part_of); gat(L.run_by);
  gcp(L.run_in); gcp(L.vmds);
}

static inline int cdiv(int a, int b) { return (a + b - 1) / b; }

extern "C" void kernel_launch(void* const* d_in, const int* in_sizes, int n_in,
                              void* d_out, int out_size, void* d_ws, size_t ws_size,
                              hipStream_t stream) {
  (void)in_sizes; (void)out_size; (void)ws_size;

  // ---- raw inputs ----
  const float* x_dc   = (const float*)d_in[0];
  const float* x_host = (const float*)d_in[1];
  const float* x_vm   = (const float*)d_in[2];
  const float* x_task = (const float*)d_in[3];
  const float* x_inst = (const float*)d_in[4];
  const int* ei_dsho = (const int*)d_in[5];
  const int* ei_dsvm = (const int*)d_in[6];
  const int* ei_hods = (const int*)d_in[7];
  const int* ei_vmds = (const int*)d_in[8];
  const int* ei_dep  = (const int*)d_in[9];
  const int* ei_part = (const int*)d_in[10];
  const int* ei_runi = (const int*)d_in[11];
  const int* ei_runb = (const int*)d_in[12];

  // ---- params (per-leaf if harness flattens the pytree, else single blob) ----
  ParamCursor pc{d_in, 13, (const float*)d_in[13], 0, n_in > 14};
  // top-level sorted: host_lin, inst_lin, layers, vm_lin; outlin sorted: b, w
  const float* host_lin_b = pc.next(EMB); const float* host_lin_w = pc.next(HID * EMB);
  const float* inst_lin_b = pc.next(EMB); const float* inst_lin_w = pc.next(HID * EMB);
  LayerP L0, L1;
  read_layer(pc, L0, IN_DIM);
  read_layer(pc, L1, HID);
  const float* vm_lin_b = pc.next(EMB);  const float* vm_lin_w = pc.next(HID * EMB);

  // ---- workspace carve-up (all chunks stay 32B-aligned: sizes % 8 == 0) ----
  float* wsp = (float*)d_ws; size_t off = 0;
  auto alloc = [&](size_t n) { float* q = wsp + off; off += n; return q; };
  float *A_dc = alloc((size_t)N_DC * HID),   *A_host = alloc((size_t)N_HOST * HID),
        *A_vm = alloc((size_t)N_VM * HID),   *A_task = alloc((size_t)N_TASK * HID),
        *A_inst = alloc((size_t)N_INST * HID);
  float *B_dc = alloc((size_t)N_DC * HID),   *B_host = alloc((size_t)N_HOST * HID),
        *B_vm = alloc((size_t)N_VM * HID),   *B_task = alloc((size_t)N_TASK * HID),
        *B_inst = alloc((size_t)N_INST * HID);
  float *t_dc   = alloc((size_t)N_DC * HID);
  float *t_host = alloc((size_t)N_HOST * HID), *hr_host = alloc((size_t)N_HOST * HID);
  float *t_vm   = alloc((size_t)N_VM * HID),   *hl_vm   = alloc((size_t)N_VM * HID);
  float *t_inst = alloc((size_t)N_INST * HID), *hr_inst = alloc((size_t)N_INST * HID);
  float *t_task = alloc((size_t)N_TASK * HID), *hl_task = alloc((size_t)N_TASK * HID);
  float *deg = alloc(N_TASK);
  float *e_part = alloc(E_PART), *e_runby = alloc(E_RUNBY);
  float *m_host = alloc(N_HOST), *den_host = alloc(N_HOST);
  float *m_inst = alloc(N_INST), *den_inst = alloc(N_INST);

  const float NEG_INF = -__builtin_huge_valf();

  // ---- launch helpers ----
  auto gemm = [&](const float* X, const float* Wm, const float* b1, const float* b2,
                  float* out, int N, int Din, int Dout, bool acc) {
    const int waves  = cdiv(N, 16 * TILES);
    const int blocks = cdiv(waves, WPB);
    if (Din == 16 && Dout == 8 && !acc)
      gnn_gemm_wmma<16, 8, false><<<blocks, WPB * 32, 0, stream>>>(X, Wm, b1, b2, out, N);
    else if (Din == 16 && Dout == 8 && acc)
      gnn_gemm_wmma<16, 8, true ><<<blocks, WPB * 32, 0, stream>>>(X, Wm, b1, b2, out, N);
    else if (Din == 8 && Dout == 8 && !acc)
      gnn_gemm_wmma< 8, 8, false><<<blocks, WPB * 32, 0, stream>>>(X, Wm, b1, b2, out, N);
    else if (Din == 8 && Dout == 8 && acc)
      gnn_gemm_wmma< 8, 8, true ><<<blocks, WPB * 32, 0, stream>>>(X, Wm, b1, b2, out, N);
    else
      gnn_gemm_wmma< 8, 5, false><<<blocks, WPB * 32, 0, stream>>>(X, Wm, b1, b2, out, N);
  };
  auto fill = [&](float* p, float v, int n) { gnn_fill<<<cdiv(n, 256), 256, 0, stream>>>(p, v, n); };
  auto fillb = [&](float* p, const float* b, int n) { gnn_fill_bias8<<<cdiv(n * HID, 256), 256, 0, stream>>>(p, b, n); };
  auto relu = [&](float* p, int n) { gnn_relu<<<cdiv(n, 256), 256, 0, stream>>>(p, n); };
  auto scat = [&](const float* h, const int* ei, float* acc, int E) {
    gnn_scatter_add<<<cdiv(E, 256), 256, 0, stream>>>(h, ei, ei + E, acc, E);
  };

  // ---- GCN degree normalization: depends only on ei_depend, compute once ----
  fill(deg, 0.0f, N_TASK);
  gnn_deg_count<<<cdiv(E_DEP, 256), 256, 0, stream>>>(ei_dep + E_DEP, deg, E_DEP);
  gnn_deg_inv<<<cdiv(N_TASK, 256), 256, 0, stream>>>(deg, N_TASK);

  // ---- one hetero layer ----
  auto run_layer = [&](const LayerP& P, int Din,
                       const float* c_dc, const float* c_host, const float* c_vm,
                       const float* c_task, const float* c_inst,
                       float* n_dc, float* n_host, float* n_vm, float* n_task, float* n_inst) {
    // host = graph_conv(dc->host, dsho) + gatv2(vm->host, run_by)
    gemm(c_host, P.dsho.w_root, P.dsho.b_rel, P.run_by.b, n_host, N_HOST, Din, HID, false);
    gemm(c_dc, P.dsho.w_rel, nullptr, nullptr, t_dc, N_DC, Din, HID, false);
    scat(t_dc, ei_dsho, n_host, E_DSHO);
    gemm(c_vm,   P.run_by.wl, P.run_by.bl, nullptr, hl_vm,   N_VM,   Din, HID, false);
    gemm(c_host, P.run_by.wr, P.run_by.br, nullptr, hr_host, N_HOST, Din, HID, false);
    fill(m_host, NEG_INF, N_HOST); fill(den_host, 0.0f, N_HOST);
    gnn_gat_logit<<<cdiv(E_RUNBY, 256), 256, 0, stream>>>(hl_vm, hr_host, P.run_by.att,
        ei_runb, ei_runb + E_RUNBY, e_runby, m_host, E_RUNBY);
    gnn_gat_exp<<<cdiv(E_RUNBY, 256), 256, 0, stream>>>(e_runby, m_host, den_host,
        ei_runb + E_RUNBY, E_RUNBY);
    gnn_gat_scatter<<<cdiv(E_RUNBY, 256), 256, 0, stream>>>(hl_vm, e_runby, den_host,
        ei_runb, ei_runb + E_RUNBY, n_host, E_RUNBY);

    // vm = graph_conv(dc->vm, dsvm) + graph_conv(inst->vm, run_in)
    gemm(c_vm, P.dsvm.w_root,   P.dsvm.b_rel,   nullptr, n_vm, N_VM, Din, HID, false);
    gemm(c_vm, P.run_in.w_root, P.run_in.b_rel, nullptr, n_vm, N_VM, Din, HID, true);
    gemm(c_dc, P.dsvm.w_rel, nullptr, nullptr, t_dc, N_DC, Din, HID, false);
    scat(t_dc, ei_dsvm, n_vm, E_DSVM);
    gemm(c_inst, P.run_in.w_rel, nullptr, nullptr, t_inst, N_INST, Din, HID, false);
    scat(t_inst, ei_runi, n_vm, E_RUNIN);

    // dc = graph_conv(host->dc, hods) + graph_conv(vm->dc, vmds)
    gemm(c_dc, P.hods.w_root, P.hods.b_rel, nullptr, n_dc, N_DC, Din, HID, false);
    gemm(c_dc, P.vmds.w_root, P.vmds.b_rel, nullptr, n_dc, N_DC, Din, HID, true);
    gemm(c_host, P.hods.w_rel, nullptr, nullptr, t_host, N_HOST, Din, HID, false);
    scat(t_host, ei_hods, n_dc, E_HODS);
    gemm(c_vm, P.vmds.w_rel, nullptr, nullptr, t_vm, N_VM, Din, HID, false);
    scat(t_vm, ei_vmds, n_dc, E_VMDS);

    // task = gcn_conv(task, depend)  (deg/dinv precomputed)
    gemm(c_task, P.depend.w, nullptr, nullptr, t_task, N_TASK, Din, HID, false);
    fillb(n_task, P.depend.b, N_TASK);
    gnn_gcn_scatter<<<cdiv(E_DEP, 256), 256, 0, stream>>>(t_task, deg,
        ei_dep, ei_dep + E_DEP, n_task, E_DEP);

    // inst = gatv2(task->inst, part_of)
    gemm(c_task, P.part_of.wl, P.part_of.bl, nullptr, hl_task, N_TASK, Din, HID, false);
    gemm(c_inst, P.part_of.wr, P.part_of.br, nullptr, hr_inst, N_INST, Din, HID, false);
    fill(m_inst, NEG_INF, N_INST); fill(den_inst, 0.0f, N_INST);
    fillb(n_inst, P.part_of.b, N_INST);
    gnn_gat_logit<<<cdiv(E_PART, 256), 256, 0, stream>>>(hl_task, hr_inst, P.part_of.att,
        ei_part, ei_part + E_PART, e_part, m_inst, E_PART);
    gnn_gat_exp<<<cdiv(E_PART, 256), 256, 0, stream>>>(e_part, m_inst, den_inst,
        ei_part + E_PART, E_PART);
    gnn_gat_scatter<<<cdiv(E_PART, 256), 256, 0, stream>>>(hl_task, e_part, den_inst,
        ei_part, ei_part + E_PART, n_inst, E_PART);

    // relu
    relu(n_dc, N_DC * HID); relu(n_host, N_HOST * HID); relu(n_vm, N_VM * HID);
    relu(n_task, N_TASK * HID); relu(n_inst, N_INST * HID);
  };

  // ---- layer 1 (reads original 16-dim features), layer 2 (8-dim) ----
  run_layer(L0, IN_DIM, x_dc, x_host, x_vm, x_task, x_inst,
            A_dc, A_host, A_vm, A_task, A_inst);
  run_layer(L1, HID, A_dc, A_host, A_vm, A_task, A_inst,
            B_dc, B_host, B_vm, B_task, B_inst);

  // ---- output linears (concat order: host, vm, inst) ----
  float* out = (float*)d_out;
  gemm(B_host, host_lin_w, host_lin_b, nullptr, out,                                 N_HOST, HID, EMB, false);
  gemm(B_vm,   vm_lin_w,   vm_lin_b,   nullptr, out + (size_t)N_HOST * EMB,          N_VM,   HID, EMB, false);
  gemm(B_inst, inst_lin_w, inst_lin_b, nullptr, out + (size_t)(N_HOST + N_VM) * EMB, N_INST, HID, EMB, false);
}
